// MultiHeadedAttention_16131897164394
// MI455X (gfx1250) — matrix operand order repaired and checked by reading, not compiled, not run on hardware
//
#include <hip/hip_runtime.h>
#include <hip/hip_bf16.h>

// ---------------------------------------------------------------------------
// Problem constants (compile-time so index math is shifts/masks, not int-div)
// ---------------------------------------------------------------------------
#define SEQ   2048   // S
#define DMODEL 1024  // D
#define NHEAD 16     // H
#define DK    64     // d_k

// ---------------------------------------------------------------------------
// Types for CDNA5 WMMA (wave32, 16x16x32 bf16 -> f32)
// ---------------------------------------------------------------------------
typedef __attribute__((ext_vector_type(8)))  float          v8f;
typedef __attribute__((ext_vector_type(16))) __bf16         bf16x16;
typedef __attribute__((ext_vector_type(16))) unsigned short u16x16;
typedef __attribute__((ext_vector_type(8)))  unsigned short u16x8;

// ---------------------------------------------------------------------------
// CDNA5 async Global->LDS copies (ASYNCcnt-tracked), with safe fallback.
// clang signature: (v4i addrspace(1)* src, v4i addrspace(3)* dst, Ii off, Ii cpol)
// ---------------------------------------------------------------------------
#if __has_builtin(__builtin_amdgcn_global_load_async_to_lds_b128)
#define HAVE_ASYNC_LDS 1
typedef int v4i __attribute__((vector_size(16)));
typedef __attribute__((address_space(1))) v4i gv4i;
typedef __attribute__((address_space(3))) v4i lv4i;
static __device__ inline void async_copy16(const unsigned short* g, unsigned short* l) {
  __builtin_amdgcn_global_load_async_to_lds_b128((gv4i*)g, (lv4i*)l, 0, 0);
}
#define ASYNC_WAIT() asm volatile("s_wait_asynccnt 0x0" ::: "memory")
#else
#define HAVE_ASYNC_LDS 0
#define ASYNC_WAIT()
#endif

static __device__ inline v8f zero8() {
  v8f v;
  #pragma unroll
  for (int i = 0; i < 8; ++i) v[i] = 0.0f;
  return v;
}

// fp32 -> bf16 (round-to-nearest-even), raw bits
static __device__ inline unsigned short f2bf(float f) {
  unsigned int u = __builtin_bit_cast(unsigned int, f);
  unsigned int r = u + 0x7FFFu + ((u >> 16) & 1u);
  return (unsigned short)(r >> 16);
}

static __device__ inline v8f wmma_bf16(bf16x16 a, bf16x16 b, v8f c) {
  return __builtin_amdgcn_wmma_f32_16x16x32_bf16(
      /*neg_a=*/false, a, /*neg_b=*/false, b,
      /*c_mod=*/(short)0, c, /*reuse_a=*/false, /*reuse_b=*/false);
}

// A-matrix fragment (16x32 bf16, row-major, leading dim ld):
// lanes 0-15: row = lane, K-chunks {0..7, 16..23};
// lanes 16-31: row = lane-16, K-chunks {8..15, 24..31}.
static __device__ inline bf16x16 load_frag_a(const unsigned short* base, int ld, int lane) {
  const unsigned short* p = base + (size_t)(lane & 15) * ld + ((lane < 16) ? 0 : 8);
  u16x8 lo = *(const u16x8*)p;
  u16x8 hi = *(const u16x8*)(p + 16);
  u16x16 v = __builtin_shufflevector(lo, hi, 0,1,2,3,4,5,6,7,8,9,10,11,12,13,14,15);
  return __builtin_bit_cast(bf16x16, v);
}

// B-matrix fragment (32x16 bf16) from [N,K] row-major (NT GEMM):
// lane = column n (mod 16); lanes 0-15 hold K={0..7, 16..23}, lanes 16-31 K={8..15, 24..31}.
static __device__ inline bf16x16 load_frag_b(const unsigned short* base, int ld, int lane) {
  const unsigned short* p = base + (size_t)(lane & 15) * ld + ((lane < 16) ? 0 : 8);
  u16x8 lo = *(const u16x8*)p;
  u16x8 hi = *(const u16x8*)(p + 16);
  u16x16 v = __builtin_shufflevector(lo, hi, 0,1,2,3,4,5,6,7,8,9,10,11,12,13,14,15);
  return __builtin_bit_cast(bf16x16, v);
}

// ---------------------------------------------------------------------------
// fp32 -> bf16 bulk conversion
// ---------------------------------------------------------------------------
__global__ void cvt_f32_bf16(const float* __restrict__ src,
                             unsigned short* __restrict__ dst, int n) {
  int i = blockIdx.x * blockDim.x + threadIdx.x;
  int stride = gridDim.x * blockDim.x;
  for (; i < n; i += stride) dst[i] = f2bf(src[i]);
}

// ---------------------------------------------------------------------------
// Tiled WMMA GEMM-NT:  C[M,N] = A[M,K] @ B[N,K]^T + bias[N]
// 128x64 block tile, 8 waves of 32x32, K-step 32.
// Double-buffered LDS, async Global->LDS staging, one barrier per K-step.
// mode 0: bf16 Q  [B,H,S,64] scaled | 1: bf16 K [B,H,S,64]
// mode 2: bf16 V^T[B,H,64,S]        | 3: f32 row-major [M,N]
// ---------------------------------------------------------------------------
__global__ __launch_bounds__(256) void gemm_nt_wmma(
    const unsigned short* __restrict__ A,
    const unsigned short* __restrict__ Bw,
    const float* __restrict__ bias,
    void* __restrict__ Cout,
    int N, int K, int mode, float scale) {
  __shared__ __align__(16) unsigned short As[2][128 * 40];
  __shared__ __align__(16) unsigned short Bs[2][64 * 40];

  const int tid  = threadIdx.x;
  const int lane = tid & 31;
  const int wave = tid >> 5;
  const int m0 = blockIdx.y * 128;
  const int n0 = blockIdx.x * 64;
  const int wm = (wave & 3) << 5;
  const int wn = (wave >> 2) << 5;

  const int arow = tid >> 1;        // 0..127
  const int acol = (tid & 1) * 16;  // 0,16
  const int brow = tid >> 2;        // 0..63
  const int bcol = (tid & 3) * 8;   // 0,8,16,24

  v8f c00 = zero8(), c01 = zero8(), c10 = zero8(), c11 = zero8();

  auto stage = [&](int k0, int buf) {
    const unsigned short* ga = A  + (size_t)(m0 + arow) * K + k0 + acol;
    const unsigned short* gb = Bw + (size_t)(n0 + brow) * K + k0 + bcol;
#if HAVE_ASYNC_LDS
    async_copy16(ga,     &As[buf][arow * 40 + acol]);
    async_copy16(ga + 8, &As[buf][arow * 40 + acol + 8]);
    async_copy16(gb,     &Bs[buf][brow * 40 + bcol]);
#else
    *(u16x8*)&As[buf][arow * 40 + acol]     = *(const u16x8*)ga;
    *(u16x8*)&As[buf][arow * 40 + acol + 8] = *(const u16x8*)(ga + 8);
    *(u16x8*)&Bs[buf][brow * 40 + bcol]     = *(const u16x8*)gb;
#endif
  };

  stage(0, 0);
  ASYNC_WAIT();
  __syncthreads();

  int cur = 0;
  for (int k0 = 0; k0 < K; k0 += 32) {
    if (k0 + 32 < K) stage(k0 + 32, cur ^ 1);  // overlap next tile with compute

    bf16x16 a0 = load_frag_a(&As[cur][(wm +  0) * 40], 40, lane);
    bf16x16 a1 = load_frag_a(&As[cur][(wm + 16) * 40], 40, lane);
    bf16x16 b0 = load_frag_b(&Bs[cur][(wn +  0) * 40], 40, lane);
    bf16x16 b1 = load_frag_b(&Bs[cur][(wn + 16) * 40], 40, lane);
    c00 = wmma_bf16(a0, b0, c00);
    c01 = wmma_bf16(a0, b1, c01);
    c10 = wmma_bf16(a1, b0, c10);
    c11 = wmma_bf16(a1, b1, c11);

    ASYNC_WAIT();
    __syncthreads();
    cur ^= 1;
  }

  // epilogue: C layout -> VGPR r holds row M = r + (lane<16?0:8)
  const int hb = (lane < 16) ? 0 : 8;
  const int nl = lane & 15;
  #pragma unroll
  for (int tm = 0; tm < 2; ++tm) {
    #pragma unroll
    for (int tn = 0; tn < 2; ++tn) {
      v8f c = (tm == 0) ? ((tn == 0) ? c00 : c01) : ((tn == 0) ? c10 : c11);
      #pragma unroll
      for (int r = 0; r < 8; ++r) {
        int m = m0 + wm + tm * 16 + r + hb;
        int n = n0 + wn + tn * 16 + nl;
        float v = c[r] + bias[n];
        if (mode == 3) {
          ((float*)Cout)[(size_t)m * N + n] = v;
        } else {
          int bb = m >> 11, ss = m & (SEQ - 1);      // SEQ = 2048
          int hh = n >> 6,  dk = n & (DK - 1);
          unsigned short bits = f2bf(mode == 0 ? v * scale : v);
          size_t idx;
          if (mode == 2) idx = (((size_t)bb * NHEAD + hh) * DK + dk) * SEQ + ss;
          else           idx = (((size_t)bb * NHEAD + hh) * SEQ + ss) * DK + dk;
          ((unsigned short*)Cout)[idx] = bits;
        }
      }
    }
  }
}

// ---------------------------------------------------------------------------
// Flash attention: block = 128 q rows (8 waves x 16), t-step = 32.
// K/V tiles staged once per block into double-buffered LDS (async),
// online softmax with cross-lane shuffles, P relayout via per-wave LDS.
// ---------------------------------------------------------------------------
__global__ __launch_bounds__(256) void flash_attn_wmma(
    const unsigned short* __restrict__ Qh,   // [B,H,S,64] bf16 (pre-scaled)
    const unsigned short* __restrict__ Kh,   // [B,H,S,64] bf16
    const unsigned short* __restrict__ Vt,   // [B,H,64,S] bf16
    const unsigned char* __restrict__ mask,  // [B,S,S] bool
    unsigned short* __restrict__ X) {        // [B,S,1024] bf16
  __shared__ __align__(16) unsigned short Ks[2][32 * 72];  // [t 0..31][d 0..63]
  __shared__ __align__(16) unsigned short Vs[2][64 * 40];  // [d 0..63][t 0..31]
  __shared__ __align__(16) unsigned short Ps[8][16 * 40];  // per-wave P tile

  const int tid = threadIdx.x, lane = tid & 31, wave = tid >> 5;
  const int b = blockIdx.z, h = blockIdx.y;
  const int q0 = blockIdx.x * 128 + wave * 16;

  const unsigned short* Qb = Qh + (((size_t)b * NHEAD + h) * SEQ) * DK;
  const unsigned short* Kb = Kh + (((size_t)b * NHEAD + h) * SEQ) * DK;
  const unsigned short* Vb = Vt + (((size_t)b * NHEAD + h) * DK) * SEQ;
  const unsigned char*  Mb = mask + (size_t)b * SEQ * SEQ;

  // Q fragments live in registers for the whole pass (K-dim = d: 0..31 / 32..63)
  bf16x16 aq0 = load_frag_a(Qb + (size_t)q0 * DK,      DK, lane);
  bf16x16 aq1 = load_frag_a(Qb + (size_t)q0 * DK + 32, DK, lane);

  v8f o0 = zero8(), o1 = zero8(), o2 = zero8(), o3 = zero8();
  float mst[8], lst[8];
  #pragma unroll
  for (int r = 0; r < 8; ++r) { mst[r] = -1e30f; lst[r] = 0.0f; }

  const int hb = (lane < 16) ? 0 : 8;
  const int nl = lane & 15;
  unsigned short* pw = &Ps[wave][0];

  // cooperative K/V tile staging: one b128 per thread per tile
  const int krow = tid >> 3, kcol = (tid & 7) * 8;  // 32 x 64
  const int vrow = tid >> 2, vcol = (tid & 3) * 8;  // 64 x 32
  auto stage = [&](int t0, int buf) {
    const unsigned short* gk = Kb + (size_t)(t0 + krow) * DK + kcol;
    const unsigned short* gv = Vb + (size_t)vrow * SEQ + t0 + vcol;
#if HAVE_ASYNC_LDS
    async_copy16(gk, &Ks[buf][krow * 72 + kcol]);
    async_copy16(gv, &Vs[buf][vrow * 40 + vcol]);
#else
    *(u16x8*)&Ks[buf][krow * 72 + kcol] = *(const u16x8*)gk;
    *(u16x8*)&Vs[buf][vrow * 40 + vcol] = *(const u16x8*)gv;
#endif
  };

  stage(0, 0);
  ASYNC_WAIT();
  __syncthreads();

  int cur = 0;
  for (int t0 = 0; t0 < SEQ; t0 += 32) {
    if (t0 + 32 < SEQ) {
      stage(t0 + 32, cur ^ 1);                                  // overlap
      if (t0 + 64 < SEQ)
        __builtin_prefetch(Kb + (size_t)(t0 + 64) * DK, 0, 1);  // global_prefetch_b8
    }

    // scores: S[16x32] = Q(16x64) @ K(t0..t0+31,:)^T   (from LDS tiles)
    v8f s0 = zero8(), s1 = zero8();
    s0 = wmma_bf16(aq0, load_frag_b(&Ks[cur][ 0 * 72 +  0], 72, lane), s0);
    s0 = wmma_bf16(aq1, load_frag_b(&Ks[cur][ 0 * 72 + 32], 72, lane), s0);
    s1 = wmma_bf16(aq0, load_frag_b(&Ks[cur][16 * 72 +  0], 72, lane), s1);
    s1 = wmma_bf16(aq1, load_frag_b(&Ks[cur][16 * 72 + 32], 72, lane), s1);

    // mask + online softmax per C-layout row slot
    #pragma unroll
    for (int r = 0; r < 8; ++r) {
      int q = q0 + r + hb;
      const unsigned char* mrow = Mb + (size_t)q * SEQ + t0;
      float v0 = mrow[nl]      ? s0[r] : -1e9f;
      float v1 = mrow[16 + nl] ? s1[r] : -1e9f;
      float tmax = fmaxf(v0, v1);
      #pragma unroll
      for (int off = 1; off < 16; off <<= 1)
        tmax = fmaxf(tmax, __shfl_xor(tmax, off, 32));
      float mnew  = fmaxf(mst[r], tmax);
      float alpha = __expf(mst[r] - mnew);
      float p0 = __expf(v0 - mnew);
      float p1 = __expf(v1 - mnew);
      float rs = p0 + p1;
      #pragma unroll
      for (int off = 1; off < 16; off <<= 1)
        rs += __shfl_xor(rs, off, 32);
      lst[r] = lst[r] * alpha + rs;
      mst[r] = mnew;
      o0[r] *= alpha; o1[r] *= alpha; o2[r] *= alpha; o3[r] *= alpha;
      pw[(r + hb) * 40 + nl]      = f2bf(p0);   // stage P tile (16x32, ld=40)
      pw[(r + hb) * 40 + 16 + nl] = f2bf(p1);
    }
    // same-wave DS store -> DS load relayout (DS in-order; make it explicit)
    asm volatile("s_wait_dscnt 0x0" ::: "memory");
    bf16x16 pa = load_frag_a(pw, 40, lane);

    // O += P(16x32) @ V[t0..t0+31,:]  (B operand from V^T tile in LDS)
    o0 = wmma_bf16(pa, load_frag_b(&Vs[cur][ 0 * 40], 40, lane), o0);
    o1 = wmma_bf16(pa, load_frag_b(&Vs[cur][16 * 40], 40, lane), o1);
    o2 = wmma_bf16(pa, load_frag_b(&Vs[cur][32 * 40], 40, lane), o2);
    o3 = wmma_bf16(pa, load_frag_b(&Vs[cur][48 * 40], 40, lane), o3);

    ASYNC_WAIT();
    __syncthreads();
    cur ^= 1;
  }

  // normalize and write X[b, q, h*64 + d] bf16 (heads concat on feature dim)
  #pragma unroll
  for (int r = 0; r < 8; ++r) {
    float inv = 1.0f / lst[r];
    int q = q0 + r + hb;
    unsigned short* xr = X + ((size_t)b * SEQ + q) * DMODEL + h * DK;
    xr[ 0 + nl] = f2bf(o0[r] * inv);
    xr[16 + nl] = f2bf(o1[r] * inv);
    xr[32 + nl] = f2bf(o2[r] * inv);
    xr[48 + nl] = f2bf(o3[r] * inv);
  }
}

// ---------------------------------------------------------------------------
// Host-side orchestration
// ---------------------------------------------------------------------------
extern "C" void kernel_launch(void* const* d_in, const int* in_sizes, int n_in,
                              void* d_out, int out_size, void* d_ws, size_t ws_size,
                              hipStream_t stream) {
  (void)in_sizes; (void)n_in; (void)out_size; (void)ws_size;
  const int B = 2, S = SEQ, D = DMODEL;
  const size_t MN = (size_t)B * S * D;  // 4M activation elements
  const size_t WN = (size_t)D * D;      // 1M weight elements

  const float* query = (const float*)d_in[0];
  const float* key   = (const float*)d_in[1];
  const float* value = (const float*)d_in[2];
  const unsigned char* mask = (const unsigned char*)d_in[3];
  const float* Wq = (const float*)d_in[4];  const float* bq = (const float*)d_in[5];
  const float* Wk = (const float*)d_in[6];  const float* bk = (const float*)d_in[7];
  const float* Wv = (const float*)d_in[8];  const float* bv = (const float*)d_in[9];
  const float* Wo = (const float*)d_in[10]; const float* bo = (const float*)d_in[11];

  unsigned short* ws  = (unsigned short*)d_ws;
  unsigned short* qb  = ws;
  unsigned short* kb  = qb  + MN;
  unsigned short* vb  = kb  + MN;
  unsigned short* wqb = vb  + MN;
  unsigned short* wkb = wqb + WN;
  unsigned short* wvb = wkb + WN;
  unsigned short* wob = wvb + WN;
  unsigned short* Qh  = wob + WN;      // [B,H,S,64], pre-scaled
  unsigned short* Kh  = Qh  + MN;      // [B,H,S,64]
  unsigned short* Vt  = Kh  + MN;      // [B,H,64,S]
  unsigned short* X   = Vt  + MN;      // [B,S,1024]

  auto cvt = [&](const float* s, unsigned short* d, size_t n) {
    int blocks = (int)((n + 1023) / 1024);
    cvt_f32_bf16<<<blocks, 256, 0, stream>>>(s, d, (int)n);
  };
  cvt(query, qb, MN);
  cvt(key,   kb, MN);
  cvt(value, vb, MN);
  cvt(Wq, wqb, WN);
  cvt(Wk, wkb, WN);
  cvt(Wv, wvb, WN);
  cvt(Wo, wob, WN);

  dim3 gg(D / 64, (B * S) / 128);  // (16, 32)
  const float scale = 0.125f;      // 1/sqrt(64)
  gemm_nt_wmma<<<gg, 256, 0, stream>>>(qb, wqb, bq, Qh, D, D, 0, scale);
  gemm_nt_wmma<<<gg, 256, 0, stream>>>(kb, wkb, bk, Kh, D, D, 1, 1.0f);
  gemm_nt_wmma<<<gg, 256, 0, stream>>>(vb, wvb, bv, Vt, D, D, 2, 1.0f);

  dim3 ga(S / 128, NHEAD, B);
  flash_attn_wmma<<<ga, 256, 0, stream>>>(Qh, Kh, Vt, mask, X);

  gemm_nt_wmma<<<gg, 256, 0, stream>>>(X, wob, bo, d_out, D, D, 3, 1.0f);
}
